// QuantizedDenseLayer_4114578669853
// MI455X (gfx1250) — 8-bit matrix operand order repaired and checked by reading, compiled, not run on hardware
//
#include <hip/hip_runtime.h>
#include <stdint.h>

typedef __attribute__((ext_vector_type(8))) int v8i;
typedef int v4i_ __attribute__((vector_size(16)));

#if defined(__has_builtin)
#if __has_builtin(__builtin_amdgcn_global_load_async_to_lds_b128)
#define USE_ASYNC_LDS 1
#endif
#endif
#ifndef USE_ASYNC_LDS
#define USE_ASYNC_LDS 0
#endif

namespace {
constexpr int NB    = 64;
constexpr int CIN   = 512;
constexpr int CMID  = 128;
constexpr int COUT  = 32;
constexpr int W28   = 28;
constexpr int NPIX  = 784;
constexpr int HP    = 30;
constexpr int PADHW = 900;
constexpr int KW2   = 1152;

constexpr size_t OFF_A1  = 0;
constexpr size_t SZ_A1   = (size_t)NB * NPIX * CIN;
constexpr size_t OFF_A2  = OFF_A1 + SZ_A1;
constexpr size_t SZ_A2   = (size_t)NB * PADHW * CMID;
constexpr size_t OFF_W1  = OFF_A2 + SZ_A2;
constexpr size_t OFF_W2  = OFF_W1 + (size_t)CMID * CIN;
constexpr size_t OFF_W1S = OFF_W2 + (size_t)COUT * KW2;
constexpr size_t OFF_W2S = OFF_W1S + CMID * sizeof(int);
}

__device__ __forceinline__ void stage_to_lds(const int8_t* __restrict__ g,
                                             int8_t* s, int nbytes, int tid) {
#if USE_ASYNC_LDS
  for (int i = tid * 16; i < nbytes; i += 256 * 16) {
    __builtin_amdgcn_global_load_async_to_lds_b128(
        (__attribute__((address_space(1))) v4i_*)(__attribute__((address_space(1))) void*)(void*)(g + i),
        (__attribute__((address_space(3))) v4i_*)(__attribute__((address_space(3))) void*)(s + i),
        0, 0);
  }
#if __has_builtin(__builtin_amdgcn_s_wait_asynccnt)
  __builtin_amdgcn_s_wait_asynccnt(0);
#else
  asm volatile("s_wait_asynccnt 0x0" ::: "memory");
#endif
#else
  for (int i = tid * 16; i < nbytes; i += 256 * 16) {
    *(int4*)(s + i) = *(const int4*)(g + i);
  }
#endif
  __syncthreads();
}

__device__ __forceinline__ v8i load_a_frag(const uint8_t* p) {
  const uint32_t* ap = (const uint32_t*)p;
  v8i av;
  av[0] = ap[0];  av[1] = ap[1];  av[2] = ap[2];  av[3] = ap[3];
  av[4] = ap[8];  av[5] = ap[9];  av[6] = ap[10]; av[7] = ap[11];
  return av;
}

__device__ __forceinline__ v8i load_b_frag_lds(const int8_t* p) {
  int4 q0 = *(const int4*)p;
  int4 q1 = *(const int4*)(p + 32);
  v8i bv;
  bv[0] = q0.x; bv[1] = q0.y; bv[2] = q0.z; bv[3] = q0.w;
  bv[4] = q1.x; bv[5] = q1.y; bv[6] = q1.z; bv[7] = q1.w;
  return bv;
}

__global__ void k_prep_w(const float* __restrict__ w1f, const float* __restrict__ w2f,
                         int8_t* __restrict__ w1q, int8_t* __restrict__ w2q,
                         int* __restrict__ w1s, int* __restrict__ w2s) {
  int t = threadIdx.x;
  if (t < CMID) {
    int s = 0;
    for (int c = 0; c < CIN; ++c) {
      int v = __float2int_rn(w1f[t * CIN + c]);
      w1q[t * CIN + c] = (int8_t)v;
      s += v;
    }
    w1s[t] = s;
  }
  if (t < COUT) {
    int s = 0;
    for (int ic = 0; ic < CMID; ++ic)
      for (int tap = 0; tap < 9; ++tap) {
        int v = __float2int_rn(w2f[(t * CMID + ic) * 9 + tap]);
        w2q[t * KW2 + tap * CMID + ic] = (int8_t)v;
        s += v;
      }
    w2s[t] = s;
  }
}

__global__ void k_bn_quant(const int* __restrict__ x, const int* __restrict__ cid,
                           const float* __restrict__ w_bn, const float* __restrict__ b_bn,
                           const float* __restrict__ z_in, const float* __restrict__ z_out,
                           const float* __restrict__ Mq, uint32_t* __restrict__ a1) {
  int idx = blockIdx.x * blockDim.x + threadIdx.x;
  int p  = idx % NPIX;
  int c4 = (idx / NPIX) & 127;
  int b  = idx / (NPIX * 128);
  int cl = cid[b];
  float zi = z_in[cl], zo = z_out[cl], m = Mq[cl];
  uint32_t pk = 0;
#pragma unroll
  for (int i = 0; i < 4; ++i) {
    int c = c4 * 4 + i;
    float q   = (float)x[((size_t)b * CIN + c) * NPIX + p];
    float acc = (q - zi) * w_bn[c] + b_bn[c];
    float v   = rintf(acc * m) + zo;
    v = fminf(fmaxf(v, zo), 255.f);
    pk |= ((uint32_t)(uint8_t)(int)v) << (8 * i);
  }
  a1[(size_t)(b * NPIX + p) * 128 + c4] = pk;
}

__global__ void k_fill_pad(const int* __restrict__ cid, const float* __restrict__ z2_in,
                           uint32_t* __restrict__ a2) {
  int idx = blockIdx.x * blockDim.x + threadIdx.x;
  int b = idx / (PADHW * CMID / 4);
  uint8_t v = (uint8_t)__float2int_rn(z2_in[cid[b]]);
  a2[idx] = v * 0x01010101u;
}

__global__ void __launch_bounds__(256) k_conv1x1(
    const uint8_t* __restrict__ a1, const int8_t* __restrict__ w1q,
    const int* __restrict__ w1s, const int* __restrict__ cid,
    const float* __restrict__ z1_in, const float* __restrict__ z1_out,
    const float* __restrict__ M1, uint8_t* __restrict__ a2) {
  __shared__ int8_t sw1[CMID * CIN];
  int tid = threadIdx.x;
  stage_to_lds(w1q, sw1, CMID * CIN, tid);

  int wv   = (blockIdx.x * 256 + tid) >> 5;
  int lane = tid & 31;
  int b = wv / 98, rem = wv % 98;
  int m0 = (rem >> 1) * 16;
  int n0 = (rem & 1) * 64;
  int lr = lane & 15, hi = lane >> 4;

  const uint8_t* arow  = a1 + ((size_t)(b * NPIX + m0 + lr) * CIN) + hi * 16;
  const int8_t*  bbase = sw1 + (size_t)(n0 + lr) * CIN + hi * 16;

  v8i acc0 = {}, acc1 = {}, acc2 = {}, acc3 = {};
#pragma unroll
  for (int k0 = 0; k0 < CIN; k0 += 64) {
    if (k0 + 64 < CIN) __builtin_prefetch(arow + k0 + 64, 0, 0);
    v8i av = load_a_frag(arow + k0);
    v8i b0 = load_b_frag_lds(bbase + 0 * 16 * CIN + k0);
    v8i b1 = load_b_frag_lds(bbase + 1 * 16 * CIN + k0);
    v8i b2 = load_b_frag_lds(bbase + 2 * 16 * CIN + k0);
    v8i b3 = load_b_frag_lds(bbase + 3 * 16 * CIN + k0);
    acc0 = __builtin_amdgcn_wmma_i32_16x16x64_iu8(false, av, true, b0, acc0, false, false);
    acc1 = __builtin_amdgcn_wmma_i32_16x16x64_iu8(false, av, true, b1, acc1, false, false);
    acc2 = __builtin_amdgcn_wmma_i32_16x16x64_iu8(false, av, true, b2, acc2, false, false);
    acc3 = __builtin_amdgcn_wmma_i32_16x16x64_iu8(false, av, true, b3, acc3, false, false);
  }

  int   cl = cid[b];
  float zo = z1_out[cl], m = M1[cl];
  int   zi = __float2int_rn(z1_in[cl]);
  auto epi = [&](v8i acc, int oc) {
    int corr = zi * w1s[oc];
#pragma unroll
    for (int g = 0; g < 8; ++g) {
      int p = m0 + g + hi * 8;
      float v = rintf((float)(acc[g] - corr) * m) + zo;
      v = fminf(fmaxf(v, 0.f), 255.f);
      int py = p / W28 + 1, px = p % W28 + 1;
      a2[((size_t)b * PADHW + py * HP + px) * CMID + oc] = (uint8_t)(int)v;
    }
  };
  epi(acc0, n0 + 0 + lr);
  epi(acc1, n0 + 16 + lr);
  epi(acc2, n0 + 32 + lr);
  epi(acc3, n0 + 48 + lr);
}

__global__ void __launch_bounds__(256) k_conv3x3(
    const uint8_t* __restrict__ a2, const int8_t* __restrict__ w2q,
    const int* __restrict__ w2s, const int* __restrict__ cid,
    const float* __restrict__ z2_in, const float* __restrict__ z2_out,
    const float* __restrict__ M2, float* __restrict__ out) {
  __shared__ int8_t sw2[COUT * KW2];
  int tid = threadIdx.x;
  stage_to_lds(w2q, sw2, COUT * KW2, tid);

  int wv   = (blockIdx.x * 256 + tid) >> 5;
  int lane = tid & 31;
  int b = wv / 49, mt = wv % 49;
  int m0 = mt * 16;
  int lr = lane & 15, hi = lane >> 4;

  int p = m0 + lr;
  int y = p / W28, x = p % W28;
  const uint8_t* abase = a2 + (size_t)b * PADHW * CMID + hi * 16;
  const int8_t*  bbase = sw2 + (size_t)lr * KW2 + hi * 16;

  v8i acc0 = {}, acc1 = {};
#pragma unroll
  for (int ks = 0; ks < 18; ++ks) {
    int tap = ks >> 1, icb = (ks & 1) << 6;
    int dy = tap / 3, dx = tap % 3;
    const uint8_t* ap = abase + (size_t)((y + dy) * HP + (x + dx)) * CMID + icb;
    __builtin_prefetch(ap + 128, 0, 0);
    v8i av = load_a_frag(ap);
    v8i b0 = load_b_frag_lds(bbase + tap * CMID + icb);
    v8i b1 = load_b_frag_lds(bbase + 16 * KW2 + tap * CMID + icb);
    acc0 = __builtin_amdgcn_wmma_i32_16x16x64_iu8(false, av, true, b0, acc0, false, false);
    acc1 = __builtin_amdgcn_wmma_i32_16x16x64_iu8(false, av, true, b1, acc1, false, false);
  }

  int   cl = cid[b];
  float zo = z2_out[cl], m = M2[cl];
  int   zi = __float2int_rn(z2_in[cl]);
  auto epi = [&](v8i acc, int oc) {
    int corr = zi * w2s[oc];
#pragma unroll
    for (int g = 0; g < 8; ++g) {
      int pp = m0 + g + hi * 8;
      float v = rintf((float)(acc[g] - corr) * m) + zo;
      v = fminf(fmaxf(v, 0.f), 255.f);
      out[((size_t)b * COUT + oc) * NPIX + pp] = v;
    }
  };
  epi(acc0, lr);
  epi(acc1, lr + 16);
}

extern "C" void kernel_launch(void* const* d_in, const int* in_sizes, int n_in,
                              void* d_out, int out_size, void* d_ws, size_t ws_size,
                              hipStream_t stream) {
  const int*   x       = (const int*)d_in[0];
  const int*   cid     = (const int*)d_in[1];
  const float* w_bn    = (const float*)d_in[2];
  const float* b_bn    = (const float*)d_in[3];
  const float* z_bn_in = (const float*)d_in[4];
  const float* z_bn_out= (const float*)d_in[5];
  const float* M_bn    = (const float*)d_in[6];
  const float* w1f     = (const float*)d_in[7];
  const float* z1_in   = (const float*)d_in[8];
  const float* z1_out  = (const float*)d_in[9];
  const float* M1      = (const float*)d_in[10];
  const float* w2f     = (const float*)d_in[11];
  const float* z2_in   = (const float*)d_in[12];
  const float* z2_out  = (const float*)d_in[13];
  const float* M2      = (const float*)d_in[14];
  float* out = (float*)d_out;

  uint8_t* ws  = (uint8_t*)d_ws;
  uint8_t* a1  = ws + OFF_A1;
  uint8_t* a2  = ws + OFF_A2;
  int8_t*  w1q = (int8_t*)(ws + OFF_W1);
  int8_t*  w2q = (int8_t*)(ws + OFF_W2);
  int*     w1s = (int*)(ws + OFF_W1S);
  int*     w2s = (int*)(ws + OFF_W2S);

  k_prep_w<<<1, 128, 0, stream>>>(w1f, w2f, w1q, w2q, w1s, w2s);

  k_bn_quant<<<(NB * NPIX * 128) / 256, 256, 0, stream>>>(
      x, cid, w_bn, b_bn, z_bn_in, z_bn_out, M_bn, (uint32_t*)a1);

  k_fill_pad<<<(NB * PADHW * CMID / 4) / 256, 256, 0, stream>>>(
      cid, z2_in, (uint32_t*)a2);

  k_conv1x1<<<784, 256, 0, stream>>>(a1, w1q, w1s, cid, z1_in, z1_out, M1, a2);

  k_conv3x3<<<392, 256, 0, stream>>>(a2, w2q, w2s, cid, z2_in, z2_out, M2, out);
}
